// RecurrentLIFNode_56831007261288
// MI455X (gfx1250) — matrix operand order repaired and checked by reading, not compiled, not run on hardware
//
#include <hip/hip_runtime.h>
#include <hip/hip_bf16.h>

// ---------------------------------------------------------------------------
// Recurrent LIF SNN on MI455X (gfx1250), fused per-(t,layer) step kernels.
//  - bf16 hi/lo split WMMA (bf16x3) for ~fp32 accuracy
//  - weights pre-split to bf16 hi/lo once per call (L2-resident, 4 MB)
//  - double-buffered LDS staging; weight tiles via async global->LDS
//    (GLOBAL_LOAD_ASYNC_TO_LDS_B128), drained with s_wait_asynccnt
//   T=128, B=256, D=H=512, L=2, TAU=2 (decay 0.5), V_TH=1.
// ---------------------------------------------------------------------------

typedef __attribute__((ext_vector_type(16))) __bf16 v16bf;
typedef __attribute__((ext_vector_type(8)))  __bf16 v8bf;
typedef __attribute__((ext_vector_type(4)))  __bf16 v4bf;
typedef __attribute__((ext_vector_type(8)))  float  v8f;
typedef int v4i __attribute__((vector_size(16)));   // builtin's b128 unit

#define GLOBAL_AS __attribute__((address_space(1)))
#define LDS_AS    __attribute__((address_space(3)))

#define SNN_T 128
#define SNN_B 256
#define SNN_D 512
#define SNN_H 512
#define SNN_L 2
#define SNN_NK (SNN_D / 32)
#define LDS_STR 40   // halfs per row: 80B pitch -> 16B aligned, conflict-free

#if defined(__gfx1250__) && __has_builtin(__builtin_amdgcn_global_load_async_to_lds_b128)
#define USE_ASYNC_LDS 1
#endif

// Convert 8 consecutive fp32 -> bf16 hi/lo and store 16B vectors to LDS.
__device__ __forceinline__ void stage8(__bf16* __restrict__ dhi,
                                       __bf16* __restrict__ dlo,
                                       const float* __restrict__ src) {
  float4 v0 = *(const float4*)(src);
  float4 v1 = *(const float4*)(src + 4);
  float f[8] = {v0.x, v0.y, v0.z, v0.w, v1.x, v1.y, v1.z, v1.w};
  v8bf hi, lo;
#pragma unroll
  for (int j = 0; j < 8; ++j) {
    __bf16 h = (__bf16)f[j];
    hi[j] = h;
    lo[j] = (__bf16)(f[j] - (float)h);
  }
  *(v8bf*)dhi = hi;
  *(v8bf*)dlo = lo;
}

// A fragment (16x32 bf16). ISA 7.12.2: lane (m = lane&15) holds K=k0..k0+7 and
// K=16+k0..16+k0+7 with k0 = 8*(lane>=16).
__device__ __forceinline__ v16bf load_a_frag(const __bf16 (*buf)[LDS_STR],
                                             int lane, int rowbase) {
  int m  = lane & 15;
  int k0 = (lane >> 4) * 8;
  const __bf16* row = buf[rowbase + m];
  v8bf lo8 = *(const v8bf*)(row + k0);
  v8bf hi8 = *(const v8bf*)(row + 16 + k0);
  v16bf r;
#pragma unroll
  for (int j = 0; j < 8; ++j) { r[j] = lo8[j]; r[8 + j] = hi8[j]; }
  return r;
}

// B fragment (32x16 bf16, B[k][n] = W[h_n][k]). Lane n holds K=0..7 and
// K=16..23, lane n+16 holds K=8..15 and K=24..31.
__device__ __forceinline__ v16bf load_b_frag(const __bf16 (*buf)[LDS_STR],
                                             int lane, int colbase) {
  int n  = lane & 15;
  int kr = (lane >> 4) * 8;
  const __bf16* row = buf[colbase + n] + kr;
  v8bf a = *(const v8bf*)(row);
  v8bf b = *(const v8bf*)(row + 16);
  v16bf r;
#pragma unroll
  for (int j = 0; j < 8; ++j) { r[j] = a[j]; r[8 + j] = b[j]; }
  return r;
}

__device__ __forceinline__ v8f wmma_bf16(v16bf a, v16bf b, v8f c) {
  return __builtin_amdgcn_wmma_f32_16x16x32_bf16(false, a, false, b,
                                                 (short)0, c, false, false);
}

// Pre-split one fp32 weight plane into bf16 hi/lo planes (run once per call).
__global__ void snn_wconv(const float* __restrict__ src,
                          __bf16* __restrict__ hi, __bf16* __restrict__ lo,
                          int n4) {
  int i = blockIdx.x * blockDim.x + threadIdx.x;
  if (i >= n4) return;
  float4 v = ((const float4*)src)[i];
  float f[4] = {v.x, v.y, v.z, v.w};
  v4bf h4, l4;
#pragma unroll
  for (int j = 0; j < 4; ++j) {
    __bf16 h = (__bf16)f[j];
    h4[j] = h;
    l4[j] = (__bf16)(f[j] - (float)h);
  }
  ((v4bf*)hi)[i] = h4;
  ((v4bf*)lo)[i] = l4;
}

__global__ void snn_zero(float* __restrict__ p, int n) {
  int i = blockIdx.x * blockDim.x + threadIdx.x;
  if (i < n) p[i] = 0.0f;
}

// One fused SNN step for one layer. Grid: 32 blocks (4 B-tiles x 8 H-tiles of
// 64x64), 256 threads (8 waves: 4 M-groups x 2 N-groups, 16x32 per wave).
__global__ __launch_bounds__(256, 2)
void snn_step(const float* __restrict__ xin,       // [B, 512] fp32 activations
              const __bf16* __restrict__ wfHi,     // [H, D] this layer
              const __bf16* __restrict__ wfLo,
              const __bf16* __restrict__ wiHi,
              const __bf16* __restrict__ wiLo,
              const float* __restrict__ bfv,       // [H]
              const float* __restrict__ biv,       // [H]
              float* __restrict__ h_state,         // [B, H] this layer
              float* __restrict__ v_state,         // [B, H] this layer
              float* __restrict__ act_out,         // layer0: next-layer input
              float* __restrict__ spike_out,       // last layer: out + t*B*H
              float* __restrict__ hT_out)          // t==T-1: out tail
{
  __shared__ __bf16 sA[2][2][64][LDS_STR];         // [buf][hi/lo][row][k]
  __shared__ __bf16 sW[2][2][2][64][LDS_STR];      // [buf][mat][hi/lo][hrow][k]

  const int tid  = threadIdx.x;
  const int lane = tid & 31;
  const int w    = tid >> 5;
  const int wm   = w & 3;                          // wave row group (16 rows)
  const int wn   = w >> 2;                         // wave col group (32 cols)
  const int bt   = blockIdx.x & 3;
  const int ht   = blockIdx.x >> 2;
  const int row0 = bt * 64;
  const int col0 = ht * 64;

  // A staging map: thread -> (row 0..63, 8-float col group); converts fp32.
  const int lr = tid >> 2;
  const int lc = (tid & 3) * 8;
  const float* aSrc = xin + (size_t)(row0 + lr) * SNN_D + lc;

  // W staging map: thread -> (tile 0..3 = fHi,fLo,iHi,iLo ; row 0..63);
  // each thread copies one 32-half (64B) row slice per chunk.
  const int wt = tid >> 6;
  const int wr = tid & 63;
  const __bf16* wSrc =
      (wt == 0 ? wfHi : wt == 1 ? wfLo : wt == 2 ? wiHi : wiLo) +
      (size_t)(col0 + wr) * SNN_D;

  auto stageA = [&](int buf, int K0) {
    stage8(&sA[buf][0][lr][lc], &sA[buf][1][lr][lc], aSrc + K0);
  };
  auto stageW = [&](int buf, int K0) {
    const __bf16* s = wSrc + K0;
    __bf16* d = &sW[buf][wt >> 1][wt & 1][wr][0];
#ifdef USE_ASYNC_LDS
#pragma unroll
    for (int q = 0; q < 4; ++q)
      __builtin_amdgcn_global_load_async_to_lds_b128(
          (GLOBAL_AS v4i*)(s + q * 8),     // global source (b128 unit)
          (LDS_AS v4i*)(d + q * 8),        // LDS destination
          /*offset=*/0, /*cpol=*/0);
#else
#pragma unroll
    for (int q = 0; q < 4; ++q)
      ((float4*)d)[q] = ((const float4*)s)[q];
#endif
  };

  v8f accF[2] = {};
  v8f accC[2] = {};

  stageA(0, 0);
  stageW(0, 0);

  for (int kc = 0; kc < SNN_NK; ++kc) {
#ifdef USE_ASYNC_LDS
    asm volatile("s_wait_asynccnt 0x0" ::: "memory");
#endif
    __syncthreads();                 // staged buf (kc&1) now visible to all
    const int cur = kc & 1;
    if (kc + 1 < SNN_NK) {           // overlap next-chunk staging with WMMA
      stageA(cur ^ 1, (kc + 1) * 32);
      stageW(cur ^ 1, (kc + 1) * 32);
      if (kc + 2 < SNN_NK) __builtin_prefetch(aSrc + (kc + 2) * 32, 0, 1);
    }

    v16bf aH = load_a_frag(sA[cur][0], lane, wm * 16);
    v16bf aL = load_a_frag(sA[cur][1], lane, wm * 16);
#pragma unroll
    for (int nt = 0; nt < 2; ++nt) {
      const int cb = wn * 32 + nt * 16;
      v16bf bH = load_b_frag(sW[cur][0][0], lane, cb);
      v16bf bL = load_b_frag(sW[cur][0][1], lane, cb);
      accF[nt] = wmma_bf16(aH, bH, accF[nt]);   // hi*hi
      accF[nt] = wmma_bf16(aH, bL, accF[nt]);   // hi*lo
      accF[nt] = wmma_bf16(aL, bH, accF[nt]);   // lo*hi
      bH = load_b_frag(sW[cur][1][0], lane, cb);
      bL = load_b_frag(sW[cur][1][1], lane, cb);
      accC[nt] = wmma_bf16(aH, bH, accC[nt]);
      accC[nt] = wmma_bf16(aH, bL, accC[nt]);
      accC[nt] = wmma_bf16(aL, bH, accC[nt]);
    }
  }

  // Epilogue: C/D layout — vgpr r, lane l: M = r + 8*(l>=16), N = l&15.
  const int m0 = (lane >> 4) * 8;
  const int n  = lane & 15;
#pragma unroll
  for (int nt = 0; nt < 2; ++nt) {
    const int hcol = col0 + wn * 32 + nt * 16 + n;
    const float bfb = bfv[hcol];
    const float bib = biv[hcol];
#pragma unroll
    for (int r = 0; r < 8; ++r) {
      const int brow = row0 + wm * 16 + m0 + r;
      const size_t idx = (size_t)brow * SNN_H + hcol;
      float fpre = accF[nt][r] + bfb;
      float cpre = accC[nt][r] + bib;
      float F = 1.0f / (1.0f + __expf(-fpre));
      float C = cpre > 0.0f ? cpre : 0.0f;
      float hp = h_state[idx];
      float vp = v_state[idx];
      float hid = F * hp + (1.0f - F) * C;
      float vn  = vp + (hid - vp) * 0.5f;          // TAU = 2
      float sp  = (vn >= 1.0f) ? 1.0f : 0.0f;      // heaviside(v - V_TH)
      vn -= sp;                                    // soft reset (V_TH = 1)
      h_state[idx] = hid;
      v_state[idx] = vn;
      if (act_out)   act_out[idx]   = hid;
      if (spike_out) spike_out[idx] = sp;
      if (hT_out)    hT_out[idx]    = hid;
    }
  }
}

extern "C" void kernel_launch(void* const* d_in, const int* in_sizes, int n_in,
                              void* d_out, int out_size, void* d_ws, size_t ws_size,
                              hipStream_t stream) {
  (void)in_sizes; (void)n_in; (void)out_size; (void)ws_size;
  const float* x  = (const float*)d_in[0];   // [T,B,D]
  const float* Wf = (const float*)d_in[1];   // [L,H,D]
  const float* bf = (const float*)d_in[2];   // [L,H]
  const float* Wi = (const float*)d_in[3];   // [L,H,D]
  const float* bi = (const float*)d_in[4];   // [L,H]
  float* out = (float*)d_out;                // spikes [T,B,H] ++ hT [L,B,H]

  const size_t BH  = (size_t)SNN_B * SNN_H;
  const size_t LHD = (size_t)SNN_L * SNN_H * SNN_D;

  // Workspace layout (fp32 then bf16 planes): ~6.6 MB total.
  float* h_state = (float*)d_ws;             // [L,B,H]
  float* v_state = h_state + SNN_L * BH;     // [L,B,H]
  float* act     = v_state + SNN_L * BH;     // [B,H] layer0 -> layer1 buffer
  __bf16* whi    = (__bf16*)(act + BH);      // [2 mats][L,H,D] hi plane
  __bf16* wlo    = whi + 2 * LHD;            // [2 mats][L,H,D] lo plane

  const int nz = 2 * SNN_L * (int)BH;        // zero h and v states
  snn_zero<<<(nz + 255) / 256, 256, 0, stream>>>(h_state, nz);

  // Pre-split weights to bf16 hi/lo once per call (graph-captured).
  const int n4 = (int)(LHD / 4);
  snn_wconv<<<(n4 + 255) / 256, 256, 0, stream>>>(Wf, whi, wlo, n4);
  snn_wconv<<<(n4 + 255) / 256, 256, 0, stream>>>(Wi, whi + LHD, wlo + LHD, n4);

  float* spikes = out;
  float* hT     = out + (size_t)SNN_T * BH;
  const size_t HD = (size_t)SNN_H * SNN_D;

  for (int t = 0; t < SNN_T; ++t) {
    const bool last = (t == SNN_T - 1);
    // layer 0: input = x[t], writes act (input for layer 1)
    snn_step<<<32, 256, 0, stream>>>(
        x + (size_t)t * SNN_B * SNN_D,
        whi, wlo,                       // Wf layer0 hi/lo
        whi + LHD, wlo + LHD,           // Wi layer0 hi/lo
        bf, bi,
        h_state, v_state,
        act, nullptr,
        last ? hT : nullptr);
    // layer 1: input = act, writes last-layer spikes
    snn_step<<<32, 256, 0, stream>>>(
        act,
        whi + HD, wlo + HD,             // Wf layer1 hi/lo
        whi + LHD + HD, wlo + LHD + HD, // Wi layer1 hi/lo
        bf + SNN_H, bi + SNN_H,
        h_state + BH, v_state + BH,
        nullptr, spikes + (size_t)t * BH,
        last ? hT + BH : nullptr);
  }
}
